// TransformerBlock_83915071030212
// MI455X (gfx1250) — matrix operand order repaired and checked by reading, not compiled, not run on hardware
//
#include <hip/hip_runtime.h>
#include <hip/hip_bf16.h>

// ---------------------------------------------------------------------------
// Types for CDNA5 WMMA (wave32) + TDM descriptors
// ---------------------------------------------------------------------------
typedef __attribute__((ext_vector_type(16))) __bf16 v16bf;
typedef __attribute__((ext_vector_type(8)))  float  v8f;
typedef __attribute__((ext_vector_type(4)))  unsigned tdm_g0_t;
typedef __attribute__((ext_vector_type(8)))  int      tdm_g1_t;
typedef __attribute__((ext_vector_type(4)))  int      tdm_g2_t;
typedef __attribute__((ext_vector_type(8)))  int      tdm_g4_t;

union Frag16 {           // 16 bf16 elements = 8 dwords (A or B operand)
    v16bf    v;
    unsigned u[8];
};

__device__ __forceinline__ unsigned short f2bf(float f) {
    unsigned u = __float_as_uint(f);
    unsigned r = u + 0x7FFFu + ((u >> 16) & 1u);   // round-to-nearest-even
    return (unsigned short)(r >> 16);
}

__device__ __forceinline__ unsigned lds_offset(const void* p) {
    // Flat LDS aperture: low 32 bits are the workgroup-relative LDS address.
    return (unsigned)(unsigned long long)p;
}

// ---------------------------------------------------------------------------
// Tensor Data Mover: async 2-D tile (bf16 elements) Global -> LDS.
//   D# per ISA 8.3/8.4: group0 {count=1, lds_addr, global_addr, type=2},
//   group1 {data_size=1(2B), tensor_dim0/1, tile_dim0/1, dim0 stride}.
//   Trailing groups zero (2-D tensor). Tracked by TENSORcnt.
//   This toolchain exposes the 6-arg builtin (clang-23 / therock-10.0).
// ---------------------------------------------------------------------------
__device__ __forceinline__ void tdm_load_2d_bf16(
    unsigned lds_off, const void* gptr,
    unsigned stride_elems,                 // elements between rows
    unsigned tensor_d0, unsigned tensor_d1,
    unsigned tile_d0,   unsigned tile_d1) {
    unsigned long long ga = (unsigned long long)gptr;
    tdm_g0_t g0 = { 1u,                                   // count=1 (valid)
                    lds_off,                              // lds_addr
                    (unsigned)ga,                         // global_addr[31:0]
                    (unsigned)((ga >> 32) & 0x1FFFFFFu) | 0x80000000u }; // +type=2
    tdm_g1_t g1 = { (int)(1u << 16),                      // data_size=1 (2 bytes)
                    (int)((tensor_d0 & 0xFFFFu) << 16),   // tensor_dim0[15:0]
                    (int)((tensor_d0 >> 16) | ((tensor_d1 & 0xFFFFu) << 16)),
                    (int)((tensor_d1 >> 16) | (tile_d0 << 16)),   // tile_dim0
                    (int)(tile_d1 & 0xFFFFu),             // tile_dim1 (tile_dim2=0)
                    (int)stride_elems,                    // tensor_dim0_stride[31:0]
                    0, 0 };
    tdm_g2_t gz = { 0, 0, 0, 0 };
    tdm_g4_t gz8 = { 0, 0, 0, 0, 0, 0, 0, 0 };
    __builtin_amdgcn_tensor_load_to_lds(g0, g1, gz, gz, gz8, 0);
}

// ---------------------------------------------------------------------------
// LayerNorm: fp32 [rows, 2048] -> bf16 [rows, 2048]
// ---------------------------------------------------------------------------
__global__ __launch_bounds__(256)
void ln_kernel(const float* __restrict__ x, const float* __restrict__ g,
               const float* __restrict__ bta, unsigned short* __restrict__ out) {
    const int C = 2048;
    __shared__ float red[256];
    const int row = blockIdx.x;
    const int tid = threadIdx.x;
    const float* xr = x + (size_t)row * C;

    float s = 0.f;
    for (int i = tid; i < C; i += 256) s += xr[i];
    red[tid] = s; __syncthreads();
    for (int st = 128; st > 0; st >>= 1) {
        if (tid < st) red[tid] += red[tid + st];
        __syncthreads();
    }
    const float mu = red[0] * (1.0f / C);
    __syncthreads();

    float v = 0.f;
    for (int i = tid; i < C; i += 256) { float d = xr[i] - mu; v += d * d; }
    red[tid] = v; __syncthreads();
    for (int st = 128; st > 0; st >>= 1) {
        if (tid < st) red[tid] += red[tid + st];
        __syncthreads();
    }
    const float rstd = rsqrtf(red[0] * (1.0f / C) + 1e-5f);

    unsigned short* orow = out + (size_t)row * C;
    for (int i = tid; i < C; i += 256) {
        float y = (xr[i] - mu) * rstd * g[i] + bta[i];
        orow[i] = f2bf(y);
    }
}

// ---------------------------------------------------------------------------
// Tiled WMMA GEMM:  C[M,N] = A[M,K](bf16) * W[N,K](fp32, converted)^T
//   64x64 output tile, K-step 32, 4 waves (128 thr), 1x4 WMMA tiles per wave.
//   A tile staged by TDM (async DMA); W tile streamed fp32->bf16 by VALU.
// MODE 0: bf16 store                       (QKV projection)
// MODE 1: fp32 store = res + acc (+bias)   (attn out-proj, FFN down)
// MODE 2: bf16 store = GELU(acc + bias)    (FFN up)
// ---------------------------------------------------------------------------
template <int MODE>
__global__ __launch_bounds__(128)
void gemm_bf16_kernel(const unsigned short* __restrict__ A,
                      const float* __restrict__ W,
                      int M, int N, int K,
                      const float* __restrict__ bias,
                      const float* __restrict__ res,
                      void* __restrict__ outp) {
    __shared__ __align__(16) unsigned short sA[64 * 32];
    __shared__ __align__(16) unsigned short sW[64 * 32];

    const int bm   = blockIdx.x * 64;
    const int bn   = blockIdx.y * 64;
    const int tid  = threadIdx.x;
    const int wv   = tid >> 5;          // wave id 0..3 -> 16-row strip
    const int lane = tid & 31;
    const int mloc = lane & 15;
    const int half = lane >> 4;

    v8f acc[4] = {};
    const unsigned sA_off = lds_offset(sA);
    const unsigned* sAr = (const unsigned*)sA;

    for (int k0 = 0; k0 < K; k0 += 32) {
        // ---- async DMA: A tile 64 rows x 32 bf16 (issued by wave 0) ----
        if (wv == 0) {
            tdm_load_2d_bf16(sA_off,
                             (const char*)A + ((size_t)bm * K + k0) * 2,
                             (unsigned)K, (unsigned)K, 4096u, 32u, 64u);
        }
        // ---- stage W tile: 64 rows x 32 fp32 -> bf16 (overlaps with DMA) ----
        for (int i = 0; i < 16; ++i) {
            int idx = tid + i * 128;               // row*32 + col
            int row = idx >> 5, col = idx & 31;
            sW[idx] = f2bf(W[(size_t)(bn + row) * K + k0 + col]);
        }
        if (wv == 0) __builtin_amdgcn_s_wait_tensorcnt(0);
        __syncthreads();

        // ---- A fragment (16x32, ISA layout: lane=M, K-pairs per VGPR) ----
        Frag16 a;
        const int arow = wv * 16 + mloc;
#pragma unroll
        for (int v = 0; v < 8; ++v)
            a.u[v] = sAr[arow * 16 + (v >> 2) * 8 + half * 4 + (v & 3)];

        // ---- 4 B fragments (32x16, lane=N col, K-pairs per VGPR) ----
#pragma unroll
        for (int g = 0; g < 4; ++g) {
            Frag16 b;
#pragma unroll
            for (int v = 0; v < 8; ++v) {
                unsigned lo = sW[(g * 16 + mloc) * 32 + 2 * ((v >> 2) * 8 + half * 4 + (v & 3))];
                unsigned hi = sW[(g * 16 + mloc) * 32 + 2 * ((v >> 2) * 8 + half * 4 + (v & 3)) + 1];
                b.u[v] = lo | (hi << 16);
            }
            acc[g] = __builtin_amdgcn_wmma_f32_16x16x32_bf16(
                false, a.v, false, b.v, (short)0, acc[g], false, false);
        }
        __syncthreads();
    }

    // ---- epilogue (C layout: VGPR r -> row r or r+8; lane -> col) ----
#pragma unroll
    for (int g = 0; g < 4; ++g) {
#pragma unroll
        for (int r = 0; r < 8; ++r) {
            const int mm = bm + wv * 16 + r + half * 8;
            const int nn = bn + g * 16 + mloc;
            float val = acc[g][r];
            if (MODE == 0) {
                ((unsigned short*)outp)[(size_t)mm * N + nn] = f2bf(val);
            } else if (MODE == 1) {
                if (bias) val += bias[nn];
                ((float*)outp)[(size_t)mm * N + nn] = res[(size_t)mm * N + nn] + val;
            } else {  // MODE 2: exact GeLU
                float t = val + bias[nn];
                float ge = 0.5f * t * (1.0f + erff(t * 0.70710678118654752f));
                ((unsigned short*)outp)[(size_t)mm * N + nn] = f2bf(ge);
            }
        }
    }
}

// ---------------------------------------------------------------------------
// Causal flash attention: one wave per (b*h, 16-query tile).
// qkv: bf16 [4096, 6144] (Q | K | V, head-major cols). out: bf16 [4096, 2048].
// Q/K/V tiles staged by TDM (async DMA), math via bf16 WMMA.
// ---------------------------------------------------------------------------
__global__ __launch_bounds__(32)
void attn_kernel(const unsigned short* __restrict__ qkv,
                 unsigned short* __restrict__ attn_out) {
    const int T = 2048, H = 16;
    __shared__ __align__(16) unsigned short sQ[16 * 128];
    __shared__ __align__(16) unsigned short sK[32 * 128];
    __shared__ __align__(16) unsigned short sV[32 * 128];
    __shared__ __align__(16) unsigned short sP[16 * 32];

    const int bh   = blockIdx.x;
    const int b    = bh / H;
    const int hh   = bh % H;
    const int qt   = blockIdx.y;
    const int q0   = qt * 16;
    const int lane = threadIdx.x;
    const int mloc = lane & 15;
    const int half = lane >> 4;
    const size_t rowbase = (size_t)b * T;
    const char*  qkv_c   = (const char*)qkv;

    // ---- async DMA: Q tile (16 rows x 128 bf16, row stride 6144) ----
    tdm_load_2d_bf16(lds_offset(sQ),
                     qkv_c + ((rowbase + q0) * 6144 + (size_t)hh * 128) * 2,
                     6144u, 6144u, 4096u, 128u, 16u);
    __builtin_amdgcn_s_wait_tensorcnt(0);
    __syncthreads();

    // ---- Q A-fragments: 4 chunks of 16x32 over head_dim ----
    Frag16 aq[4];
    {
        const unsigned* sQr = (const unsigned*)sQ;
#pragma unroll
        for (int kc = 0; kc < 4; ++kc)
#pragma unroll
            for (int v = 0; v < 8; ++v)
                aq[kc].u[v] =
                    sQr[mloc * 64 + kc * 16 + (v >> 2) * 8 + half * 4 + (v & 3)];
    }

    v8f  o[8] = {};
    float mrun[8], lrun[8];
#pragma unroll
    for (int r = 0; r < 8; ++r) { mrun[r] = -3.0e38f; lrun[r] = 0.f; }

    const float scale = 0.08838834764831845f;       // 1/sqrt(128)
    const int nchunks = (qt + 2) >> 1;              // keys <= q0+15 covered

    for (int c = 0; c < nchunks; ++c) {
        const int kb = c * 32;
        // ---- async DMA: K and V chunks (32 rows x 128 bf16 each) ----
        {
            size_t base = (rowbase + kb) * 6144 + (size_t)hh * 128;
            tdm_load_2d_bf16(lds_offset(sK), qkv_c + (base + 2048) * 2,
                             6144u, 6144u, 4096u, 128u, 32u);
            tdm_load_2d_bf16(lds_offset(sV), qkv_c + (base + 4096) * 2,
                             6144u, 6144u, 4096u, 128u, 32u);
            __builtin_amdgcn_s_wait_tensorcnt(0);
        }
        __syncthreads();

        // ---- S = Q K^T for two 16-key subtiles ----
        v8f s[2];
#pragma unroll
        for (int t = 0; t < 2; ++t) {
            v8f st = {};
#pragma unroll
            for (int kc = 0; kc < 4; ++kc) {
                Frag16 bk;                           // B: lane = key col, d pairs
#pragma unroll
                for (int v = 0; v < 8; ++v) {
                    unsigned lo = sK[(t * 16 + mloc) * 128 + kc * 32 + 2 * ((v >> 2) * 8 + half * 4 + (v & 3))];
                    unsigned hi = sK[(t * 16 + mloc) * 128 + kc * 32 + 2 * ((v >> 2) * 8 + half * 4 + (v & 3)) + 1];
                    bk.u[v] = lo | (hi << 16);
                }
                st = __builtin_amdgcn_wmma_f32_16x16x32_bf16(
                    false, aq[kc].v, false, bk.v, (short)0, st, false, false);
            }
            s[t] = st;
        }

        // ---- scale + causal mask ----
#pragma unroll
        for (int t = 0; t < 2; ++t)
#pragma unroll
            for (int r = 0; r < 8; ++r) {
                int key  = kb + t * 16 + mloc;
                int qrow = q0 + r + half * 8;
                float v = s[t][r] * scale;
                s[t][r] = (key > qrow) ? -1.0e30f : v;
            }

        // ---- online softmax (row = 16 lanes within half) ----
#pragma unroll
        for (int r = 0; r < 8; ++r) {
            float mx = fmaxf(s[0][r], s[1][r]);
            mx = fmaxf(mx, __shfl_xor(mx, 8, 32));
            mx = fmaxf(mx, __shfl_xor(mx, 4, 32));
            mx = fmaxf(mx, __shfl_xor(mx, 2, 32));
            mx = fmaxf(mx, __shfl_xor(mx, 1, 32));
            float mnew  = fmaxf(mrun[r], mx);
            float alpha = __expf(mrun[r] - mnew);
            float p0 = __expf(s[0][r] - mnew);
            float p1 = __expf(s[1][r] - mnew);
            float rs = p0 + p1;
            rs += __shfl_xor(rs, 8, 32);
            rs += __shfl_xor(rs, 4, 32);
            rs += __shfl_xor(rs, 2, 32);
            rs += __shfl_xor(rs, 1, 32);
            lrun[r] = lrun[r] * alpha + rs;
            mrun[r] = mnew;
            s[0][r] = p0; s[1][r] = p1;
#pragma unroll
            for (int g = 0; g < 8; ++g) o[g][r] *= alpha;
        }

        // ---- P (C layout) -> bf16 in LDS -> A layout fragment ----
#pragma unroll
        for (int t = 0; t < 2; ++t)
#pragma unroll
            for (int r = 0; r < 8; ++r)
                sP[(r + half * 8) * 32 + t * 16 + mloc] = f2bf(s[t][r]);

        Frag16 ap;
        {
            const unsigned* sPr = (const unsigned*)sP;
#pragma unroll
            for (int v = 0; v < 8; ++v)
                ap.u[v] = sPr[mloc * 16 + (v >> 2) * 8 + half * 4 + (v & 3)];
        }

        // ---- O += P * V  (B: lane = d col, key pairs per VGPR) ----
        {
            const unsigned* sVr = (const unsigned*)sV;
#pragma unroll
            for (int g = 0; g < 8; ++g) {
                Frag16 bv;
#pragma unroll
                for (int v = 0; v < 8; ++v)
                    bv.u[v] = (unsigned)sV[((v >> 2) * 16 + half * 8 + 2 * (v & 3)) * 128 + g * 16 + mloc] | ((unsigned)sV[((v >> 2) * 16 + half * 8 + 2 * (v & 3) + 1) * 128 + g * 16 + mloc] << 16);
                o[g] = __builtin_amdgcn_wmma_f32_16x16x32_bf16(
                    false, ap.v, false, bv.v, (short)0, o[g], false, false);
            }
        }
        __syncthreads();
    }

    // ---- normalize and store bf16 [4096, 2048] ----
#pragma unroll
    for (int g = 0; g < 8; ++g)
#pragma unroll
        for (int r = 0; r < 8; ++r) {
            int qrow = q0 + r + half * 8;
            float val = o[g][r] / lrun[r];
            attn_out[(rowbase + qrow) * 2048 + hh * 128 + g * 16 + mloc] = f2bf(val);
        }
}

// ---------------------------------------------------------------------------
// Orchestration
// ---------------------------------------------------------------------------
extern "C" void kernel_launch(void* const* d_in, const int* in_sizes, int n_in,
                              void* d_out, int out_size, void* d_ws, size_t ws_size,
                              hipStream_t stream) {
    const float* x      = (const float*)d_in[0];
    const float* ln1_g  = (const float*)d_in[1];
    const float* ln1_b  = (const float*)d_in[2];
    const float* ln2_g  = (const float*)d_in[3];
    const float* ln2_b  = (const float*)d_in[4];
    const float* w_qkv  = (const float*)d_in[5];
    const float* w_out  = (const float*)d_in[6];
    const float* w_up   = (const float*)d_in[7];
    const float* b_up   = (const float*)d_in[8];
    const float* w_down = (const float*)d_in[9];
    const float* b_down = (const float*)d_in[10];

    char* ws = (char*)d_ws;
    // h_bf:   4096x2048 bf16 (reused for ln2 output)
    // big:    4096x8192 bf16 (qkv [4096x6144] first, then FFN hidden u)
    // attn:   4096x2048 bf16
    // x1:     4096x2048 fp32 (residual after attention)
    unsigned short* h_bf    = (unsigned short*)ws;
    unsigned short* big_bf  = (unsigned short*)(ws + (size_t)16777216);
    unsigned short* attn_bf = (unsigned short*)(ws + (size_t)16777216 + 67108864);
    float*          x1      = (float*)(ws + (size_t)16777216 + 67108864 + 16777216);

    // 1) h = LN1(x)
    ln_kernel<<<4096, 256, 0, stream>>>(x, ln1_g, ln1_b, h_bf);
    // 2) qkv = h @ w_qkv^T            [4096, 6144] bf16
    gemm_bf16_kernel<0><<<dim3(64, 96), 128, 0, stream>>>(
        h_bf, w_qkv, 4096, 6144, 2048, nullptr, nullptr, big_bf);
    // 3) causal flash attention       [4096, 2048] bf16
    attn_kernel<<<dim3(32, 128), 32, 0, stream>>>(big_bf, attn_bf);
    // 4) x1 = x + attn @ w_out^T      [4096, 2048] fp32
    gemm_bf16_kernel<1><<<dim3(64, 32), 128, 0, stream>>>(
        attn_bf, w_out, 4096, 2048, 2048, nullptr, x, x1);
    // 5) h = LN2(x1)
    ln_kernel<<<4096, 256, 0, stream>>>(x1, ln2_g, ln2_b, h_bf);
    // 6) u = GELU(h @ w_up^T + b_up)  [4096, 8192] bf16
    gemm_bf16_kernel<2><<<dim3(64, 128), 128, 0, stream>>>(
        h_bf, w_up, 4096, 8192, 2048, b_up, nullptr, big_bf);
    // 7) out = x1 + u @ w_down^T + b_down   [4096, 2048] fp32
    gemm_bf16_kernel<1><<<dim3(64, 32), 128, 0, stream>>>(
        big_bf, w_down, 4096, 2048, 8192, b_down, x1, (float*)d_out);
}
